// LocalGate_19971597927216
// MI455X (gfx1250) — compile-verified, matrix operand order checked by reading, not run on hardware
//
#include <hip/hip_runtime.h>

#define N_TOKENS   131072
#define MODEL_DIM  1024
#define NUM_EXP    64
#define TOPK       2
#define NK         (N_TOKENS * TOPK)       // 262144
#define TOK_PER_BLK 128
#define KCHUNKS    (MODEL_DIM / 32)         // 32
#define HIST_CHUNK 512
#define HIST_BLKS  (NK / HIST_CHUNK)        // 512

typedef __attribute__((ext_vector_type(16))) __bf16 v16bf;
typedef __attribute__((ext_vector_type(8)))  float  v8f;

__device__ __forceinline__ unsigned short f2bf(float f) {
    unsigned u = __float_as_uint(f);
    u += 0x7FFFu + ((u >> 16) & 1u);        // round-to-nearest-even
    return (unsigned short)(u >> 16);
}
__device__ __forceinline__ unsigned pack2bf(float a, float b) {
    return (unsigned)f2bf(a) | ((unsigned)f2bf(b) << 16);   // low half = even k
}

// ---------------------------------------------------------------------------
// Kernel 0: pre-pack wg (f32) into bf16 pairs in WMMA B-fragment order:
//           Bp[chunk][jtile][p][lane], lane holds col (lane&15), kh = lane>>4
// ---------------------------------------------------------------------------
__global__ __launch_bounds__(256) void pack_wg_kernel(
    const float* __restrict__ wg, unsigned* __restrict__ Bp)
{
    int idx   = blockIdx.x * 256 + threadIdx.x;   // 0 .. 32767
    int lane  = idx & 31;
    int p     = (idx >> 5) & 7;
    int j     = (idx >> 8) & 3;
    int chunk = idx >> 10;
    int ecol  = j * 16 + (lane & 15);
    int kh    = lane >> 4;
    int k     = chunk * 32 + (p >> 2) * 16 + kh * 8 + (p & 3) * 2;
    const float* src = wg + ecol * MODEL_DIM + k;
    Bp[idx] = pack2bf(src[0], src[1]);
}

// ---------------------------------------------------------------------------
// Kernel 1: bf16 WMMA GEMM (logits) + softmax + top-2 + probs/cw outputs.
// A fragments read straight from global (float4, no LDS, no barriers in loop);
// B fragments read pre-packed from L2-resident buffer.
// ---------------------------------------------------------------------------
__global__ __launch_bounds__(256) void moe_gate_kernel(
    const float*    __restrict__ x,        // [N_TOKENS, MODEL_DIM]
    const unsigned* __restrict__ Bp,       // [32][4][8][32] packed bf16 pairs
    float* __restrict__ probs_out,         // [N_TOKENS, NUM_EXP]
    float* __restrict__ cw_out,            // [NK]
    int*   __restrict__ topk_out)          // [NK]
{
    __shared__ float Lg[TOK_PER_BLK * 68];     // logits for softmax stage
    __shared__ float rowMax[TOK_PER_BLK];
    __shared__ float rowInv[TOK_PER_BLK];

    const int tid  = threadIdx.x;
    const int lane = tid & 31;
    const int wave = tid >> 5;
    const long tok0 = (long)blockIdx.x * TOK_PER_BLK;

    const int mrow = wave * 16 + (lane & 15);   // this lane's token row
    const int kh   = lane >> 4;                 // K-half select (ISA 16-bit A layout)
    const float* arow = x + (tok0 + mrow) * MODEL_DIM + kh * 8;

    v8f zero = {0.f, 0.f, 0.f, 0.f, 0.f, 0.f, 0.f, 0.f};
    v8f c[4];
#pragma unroll
    for (int j = 0; j < 4; ++j) c[j] = zero;

#pragma unroll 2
    for (int ch = 0; ch < KCHUNKS; ++ch) {
        // Lane's 16 A values for this chunk: k = kh*8 + {0..7} and 16+kh*8 + {0..7}
        const float4* ap = (const float4*)(arow + ch * 32);
        float4 f0 = ap[0];                 // k = kh*8 + 0..3
        float4 f1 = ap[1];                 // k = kh*8 + 4..7
        float4 f2 = ap[4];                 // k = 16 + kh*8 + 0..3
        float4 f3 = ap[5];                 // k = 16 + kh*8 + 4..7
        if (ch + 1 < KCHUNKS)              // WGP-scope global_prefetch_b8 of next chunk
            __builtin_prefetch(arow + (ch + 1) * 32, 0, 3);

        union { v16bf v; unsigned u[8]; } a;
        a.u[0] = pack2bf(f0.x, f0.y);  a.u[1] = pack2bf(f0.z, f0.w);
        a.u[2] = pack2bf(f1.x, f1.y);  a.u[3] = pack2bf(f1.z, f1.w);
        a.u[4] = pack2bf(f2.x, f2.y);  a.u[5] = pack2bf(f2.z, f2.w);
        a.u[6] = pack2bf(f3.x, f3.y);  a.u[7] = pack2bf(f3.z, f3.w);

        const unsigned* bp = Bp + ch * 1024 + lane;   // [4][8][32] for this chunk
#pragma unroll
        for (int j = 0; j < 4; ++j) {
            union { v16bf v; unsigned u[8]; } b;
#pragma unroll
            for (int p = 0; p < 8; ++p) b.u[p] = bp[j * 256 + p * 32];
            c[j] = __builtin_amdgcn_wmma_f32_16x16x32_bf16(
                false, a.v, false, b.v, (short)0, c[j], false, false);
        }
    }

    // Spill accumulators per ISA C/D layout: VGPR r -> M=r (lanes 0-15) / M=r+8
#pragma unroll
    for (int j = 0; j < 4; ++j) {
#pragma unroll
        for (int r = 0; r < 8; ++r) {
            int row = wave * 16 + ((lane < 16) ? r : r + 8);
            int col = j * 16 + (lane & 15);
            Lg[row * 68 + col] = c[j][r];
        }
    }
    __syncthreads();

    // Per-token softmax stats + top-2 (ties -> lower index, matching lax.top_k)
    if (tid < TOK_PER_BLK) {
        const float* row = &Lg[tid * 68];
        float m = row[0];
#pragma unroll
        for (int e = 1; e < NUM_EXP; ++e) m = fmaxf(m, row[e]);
        float s = 0.f;
#pragma unroll
        for (int e = 0; e < NUM_EXP; ++e) s += __expf(row[e] - m);
        rowMax[tid] = m;
        rowInv[tid] = 1.f / s;

        float b1 = -3.4e38f, b2 = -3.4e38f; int i1 = 0, i2 = 0;
#pragma unroll
        for (int e = 0; e < NUM_EXP; ++e) {
            float v = row[e];
            if (v > b1)      { b2 = b1; i2 = i1; b1 = v; i1 = e; }
            else if (v > b2) { b2 = v; i2 = e; }
        }
        long t = tok0 + tid;
        topk_out[t * 2]     = i1;
        topk_out[t * 2 + 1] = i2;
        cw_out[t * 2]     = 1.0f;   // straight-through: forward value is 1.0
        cw_out[t * 2 + 1] = 1.0f;
    }
    __syncthreads();

    // Coalesced probs write: consecutive tid -> consecutive expert column
#pragma unroll
    for (int r = 0; r < (TOK_PER_BLK * NUM_EXP) / 256; ++r) {
        int idx = r * 256 + tid;
        int t = idx >> 6, e = idx & 63;
        probs_out[(tok0 + t) * NUM_EXP + e] =
            __expf(Lg[t * 68 + e] - rowMax[t]) * rowInv[t];
    }
}

// ---------------------------------------------------------------------------
// Kernel 2: per-chunk expert histogram (512 entries per single-wave block)
// ---------------------------------------------------------------------------
__global__ __launch_bounds__(32) void moe_hist_kernel(
    const int* __restrict__ idx, int* __restrict__ blockCounts)
{
    __shared__ int cnt[NUM_EXP];
    int lane = threadIdx.x;
    cnt[lane] = 0; cnt[lane + 32] = 0;
    __syncthreads();
    int base = blockIdx.x * HIST_CHUNK;
#pragma unroll
    for (int r = 0; r < HIST_CHUNK / 32; ++r)
        atomicAdd(&cnt[idx[base + r * 32 + lane]], 1);
    __syncthreads();
    blockCounts[blockIdx.x * NUM_EXP + lane]      = cnt[lane];
    blockCounts[blockIdx.x * NUM_EXP + lane + 32] = cnt[lane + 32];
}

// ---------------------------------------------------------------------------
// Kernel 3: per-expert exclusive prefix over blocks + expert offsets + splits
// ---------------------------------------------------------------------------
__global__ __launch_bounds__(64) void moe_scan_kernel(
    const int* __restrict__ blockCounts, int* __restrict__ blockBase,
    int* __restrict__ offsets, float* __restrict__ splits_out)
{
    __shared__ int tot[NUM_EXP];
    int e = threadIdx.x;
    int run = 0;
    for (int b = 0; b < HIST_BLKS; ++b) {
        int c = blockCounts[b * NUM_EXP + e];
        blockBase[b * NUM_EXP + e] = run;
        run += c;
    }
    tot[e] = run;
    splits_out[e] = (float)run;
    __syncthreads();
    if (e == 0) {
        int off = 0;
        for (int q = 0; q < NUM_EXP; ++q) { offsets[q] = off; off += tot[q]; }
    }
}

// ---------------------------------------------------------------------------
// Kernel 4: stable scatter (counting-sort placement) via intra-wave ballot match
// ---------------------------------------------------------------------------
__global__ __launch_bounds__(32) void moe_scatter_kernel(
    const int* __restrict__ idx, const int* __restrict__ blockBase,
    const int* __restrict__ offsets,
    float* __restrict__ tok_out, float* __restrict__ rev_out)
{
    __shared__ int cnt[NUM_EXP];
    int lane = threadIdx.x;
    cnt[lane]      = offsets[lane]      + blockBase[blockIdx.x * NUM_EXP + lane];
    cnt[lane + 32] = offsets[lane + 32] + blockBase[blockIdx.x * NUM_EXP + lane + 32];
    __syncthreads();

    int base = blockIdx.x * HIST_CHUNK;
    for (int r = 0; r < HIST_CHUNK / 32; ++r) {
        int i = base + r * 32 + lane;
        int e = idx[i];
        // match-any over 6 expert-id bits (wave32 ballot)
        unsigned mask = 0xFFFFFFFFu;
#pragma unroll
        for (int b = 0; b < 6; ++b) {
            unsigned bl = (unsigned)__ballot((e >> b) & 1);
            mask &= ((e >> b) & 1) ? bl : ~bl;
        }
        int leader = __ffs(mask) - 1;
        int rank   = __popc(mask & ((1u << lane) - 1u));
        int grp    = __popc(mask);
        int basep  = 0;
        if (lane == leader) { basep = cnt[e]; cnt[e] = basep + grp; }
        basep = __shfl(basep, leader, 32);
        int pos = basep + rank;
        rev_out[i]   = (float)pos;        // reversed_ordering[i] = sorted position
        tok_out[pos] = (float)(i >> 1);   // token_ordering[pos]  = slot / K
        __syncthreads();                  // order rounds (single-wave fence)
    }
}

// ---------------------------------------------------------------------------
extern "C" void kernel_launch(void* const* d_in, const int* in_sizes, int n_in,
                              void* d_out, int out_size, void* d_ws, size_t ws_size,
                              hipStream_t stream) {
    const float* x  = (const float*)d_in[0];   // [131072, 1024]
    const float* wg = (const float*)d_in[1];   // [64, 1024]

    float* out       = (float*)d_out;
    float* tok_out   = out;                         // [NK]
    float* rev_out   = out + NK;                    // [NK]
    float* cw_out    = out + 2 * NK;                // [NK]
    float* split_out = out + 3 * NK;                // [64]
    float* probs_out = out + 3 * NK + NUM_EXP;      // [N, 64]

    int* topk        = (int*)d_ws;                  // [NK]
    int* blockCounts = topk + NK;                   // [512*64]
    int* blockBase   = blockCounts + HIST_BLKS * NUM_EXP;
    int* offsets     = blockBase + HIST_BLKS * NUM_EXP;
    unsigned* Bp     = (unsigned*)(offsets + NUM_EXP);   // [32*4*8*32] = 32768

    pack_wg_kernel<<<(KCHUNKS * 4 * 8 * 32) / 256, 256, 0, stream>>>(wg, Bp);
    moe_gate_kernel<<<N_TOKENS / TOK_PER_BLK, 256, 0, stream>>>(
        x, Bp, probs_out, cw_out, topk);
    moe_hist_kernel<<<HIST_BLKS, 32, 0, stream>>>(topk, blockCounts);
    moe_scan_kernel<<<1, 64, 0, stream>>>(blockCounts, blockBase, offsets, split_out);
    moe_scatter_kernel<<<HIST_BLKS, 32, 0, stream>>>(topk, blockBase, offsets,
                                                     tok_out, rev_out);
}
